// Model_71940702208358
// MI455X (gfx1250) — compile-verified, matrix operand order checked by reading, not run on hardware
//
#include <hip/hip_runtime.h>
#include <math.h>

typedef __attribute__((ext_vector_type(16))) _Float16 v16h;
typedef __attribute__((ext_vector_type(8)))  float    v8f;

union FragU { uint4 u[2]; v16h v; };
union PackU { _Float16 h[16]; uint4 u[2]; };

#define LDSW 40   // _Float16 per LDS row = 80B stride (16B aligned, bank-spread)

// C = ACT(alpha * A@B + bias), batched over blockIdx.z via strides.
// BT=false: B is K x N row-major. BT=true: B is N x K row-major (Q@K^T case).
template<int ACT, bool BT>
__global__ __launch_bounds__(256)
void gemm_kernel(const float* __restrict__ A, int lda, long long strideA,
                 const float* __restrict__ Bm, int ldb, long long strideB,
                 const float* __restrict__ bias,
                 float* __restrict__ C, int ldc, long long strideC,
                 int M, int N, int K, float alpha)
{
    A  += (long long)blockIdx.z * strideA;
    Bm += (long long)blockIdx.z * strideB;
    C  += (long long)blockIdx.z * strideC;

    __shared__ __align__(16) _Float16 As[2][128][LDSW];
    __shared__ __align__(16) _Float16 Bs[2][128][LDSW];

    const int tid  = threadIdx.x;
    const int lane = tid & 31;           // wave32
    const int wave = tid >> 5;           // 8 waves
    const int half = lane >> 4;          // ISA 16-bit fragment half select
    const int lrow = lane & 15;
    const int wmOff = (wave & 1) * 64;   // 2 waves along M
    const int wnOff = (wave >> 1) * 32;  // 4 waves along N
    const int blockM = blockIdx.y * 128;
    const int blockN = blockIdx.x * 128;

    // staging geometry: A / BT-B: 2 threads per row, 16 cols each.  NN-B: 8 threads per k-row.
    const int arow = tid >> 1, ac0 = (tid & 1) << 4;
    const int bkc  = tid >> 3, bn0 = (tid & 7) << 4;

    float4 RA[4], RB[4];

    auto loadA = [&](int k0) {
        const int gr = blockM + arow;
        const bool fast = (blockM + 128 <= M) && (k0 + 32 <= K);
        if (fast) {
            const float* src = A + (long long)gr * lda + (k0 + ac0);
            #pragma unroll
            for (int q = 0; q < 4; ++q) RA[q] = ((const float4*)src)[q];
        } else {
            const int cr = gr < M ? gr : (M - 1);
            const float* srow = A + (long long)cr * lda;
            #pragma unroll
            for (int q = 0; q < 4; ++q) {
                float t[4];
                #pragma unroll
                for (int e = 0; e < 4; ++e) {
                    int gc = k0 + ac0 + 4 * q + e;
                    int cc = gc < K ? gc : (K - 1);
                    float v = srow[cc];                       // unconditional, clamped addr
                    t[e] = (gr < M && gc < K) ? v : 0.0f;     // v_cndmask, no branch
                }
                RA[q] = make_float4(t[0], t[1], t[2], t[3]);
            }
        }
    };

    auto loadB = [&](int k0) {
        if (BT) {
            const int gn = blockN + arow;
            const bool fast = (blockN + 128 <= N) && (k0 + 32 <= K);
            if (fast) {
                const float* src = Bm + (long long)gn * ldb + (k0 + ac0);
                #pragma unroll
                for (int q = 0; q < 4; ++q) RB[q] = ((const float4*)src)[q];
            } else {
                const int cn = gn < N ? gn : (N - 1);
                const float* srow = Bm + (long long)cn * ldb;
                #pragma unroll
                for (int q = 0; q < 4; ++q) {
                    float t[4];
                    #pragma unroll
                    for (int e = 0; e < 4; ++e) {
                        int gc = k0 + ac0 + 4 * q + e;
                        int cc = gc < K ? gc : (K - 1);
                        float v = srow[cc];
                        t[e] = (gn < N && gc < K) ? v : 0.0f;
                    }
                    RB[q] = make_float4(t[0], t[1], t[2], t[3]);
                }
            }
        } else {
            const int gk = k0 + bkc;
            const bool fast = (blockN + 128 <= N) && (k0 + 32 <= K);
            if (fast) {
                const float* src = Bm + (long long)gk * ldb + (blockN + bn0);
                #pragma unroll
                for (int q = 0; q < 4; ++q) RB[q] = ((const float4*)src)[q];
            } else {
                const int ck = gk < K ? gk : (K - 1);
                const float* srow = Bm + (long long)ck * ldb;
                #pragma unroll
                for (int q = 0; q < 4; ++q) {
                    float t[4];
                    #pragma unroll
                    for (int e = 0; e < 4; ++e) {
                        int gn = blockN + bn0 + 4 * q + e;
                        int cn = gn < N ? gn : (N - 1);
                        float v = srow[cn];
                        t[e] = (gk < K && gn < N) ? v : 0.0f;
                    }
                    RB[q] = make_float4(t[0], t[1], t[2], t[3]);
                }
            }
        }
    };

    auto storeA = [&](int buf) {
        PackU pk;
        #pragma unroll
        for (int q = 0; q < 4; ++q) {
            const float* f = (const float*)&RA[q];
            #pragma unroll
            for (int e = 0; e < 4; ++e) pk.h[4 * q + e] = (_Float16)f[e];
        }
        uint4* dst = (uint4*)&As[buf][arow][ac0];
        dst[0] = pk.u[0];
        dst[1] = pk.u[1];
    };

    auto storeB = [&](int buf) {
        if (BT) {
            PackU pk;
            #pragma unroll
            for (int q = 0; q < 4; ++q) {
                const float* f = (const float*)&RB[q];
                #pragma unroll
                for (int e = 0; e < 4; ++e) pk.h[4 * q + e] = (_Float16)f[e];
            }
            uint4* dst = (uint4*)&Bs[buf][arow][ac0];
            dst[0] = pk.u[0];
            dst[1] = pk.u[1];
        } else {
            #pragma unroll
            for (int q = 0; q < 4; ++q) {
                const float* f = (const float*)&RB[q];
                #pragma unroll
                for (int e = 0; e < 4; ++e)
                    Bs[buf][bn0 + 4 * q + e][bkc] = (_Float16)f[e];
            }
        }
    };

    v8f acc[4][2] = {};

    auto compute = [&](int buf) {
        FragU af[4], bfr[2];
        #pragma unroll
        for (int i = 0; i < 4; ++i) {
            const uint32_t* p = (const uint32_t*)&As[buf][wmOff + i * 16 + lrow][0];
            af[i].u[0] = *(const uint4*)(p + half * 4);
            af[i].u[1] = *(const uint4*)(p + 8 + half * 4);
        }
        #pragma unroll
        for (int j = 0; j < 2; ++j) {
            const uint32_t* p = (const uint32_t*)&Bs[buf][wnOff + j * 16 + lrow][0];
            bfr[j].u[0] = *(const uint4*)(p + half * 4);
            bfr[j].u[1] = *(const uint4*)(p + 8 + half * 4);
        }
        #pragma unroll
        for (int i = 0; i < 4; ++i)
            #pragma unroll
            for (int j = 0; j < 2; ++j)
                acc[i][j] = __builtin_amdgcn_wmma_f32_16x16x32_f16(
                    false, af[i].v, false, bfr[j].v, (short)0, acc[i][j], false, false);
    };

    const int nk = (K + 31) >> 5;
    loadA(0);
    loadB(0);
    storeA(0);
    storeB(0);
    __syncthreads();

    for (int t = 0; t < nk; ++t) {
        const int cur = t & 1, nxt = cur ^ 1;
        const bool more = (t + 1) < nk;
        if (more) {
            loadA((t + 1) << 5);
            loadB((t + 1) << 5);
            if ((t + 2) < nk) {
                int gr = blockM + arow;
                __builtin_prefetch(A + (long long)(gr < M ? gr : 0) * lda + ((t + 2) << 5) + ac0, 0, 0);
            }
        }
        compute(cur);
        if (more) { storeA(nxt); storeB(nxt); }
        __syncthreads();
    }

    // epilogue per ISA C layout: VGPR r -> M = r + half*8; N = lrow
    #pragma unroll
    for (int j = 0; j < 2; ++j) {
        int col = blockN + wnOff + j * 16 + lrow;
        if (col >= N) continue;
        float bv = bias ? bias[col] : 0.0f;
        #pragma unroll
        for (int i = 0; i < 4; ++i) {
            #pragma unroll
            for (int r = 0; r < 8; ++r) {
                int row = blockM + wmOff + i * 16 + half * 8 + r;
                if (row < M) {
                    float x0 = alpha * acc[i][j][r] + bv;
                    if (ACT == 1) x0 = 0.5f * x0 * (1.0f + erff(x0 * 0.70710678118654752f));
                    C[(long long)row * ldc + col] = x0;
                }
            }
        }
    }
}

__global__ __launch_bounds__(256)
void softmax_kernel(float* __restrict__ s, int rowlen, int rowstride)
{
    float* row = s + (long long)blockIdx.x * rowstride;
    __shared__ float red[256];
    int tid = threadIdx.x;
    float vals[9];
    int cnt = 0;
    float mx = -3.0e38f;
    for (int i = tid; i < rowlen; i += 256) {
        float v = row[i];
        vals[cnt++] = v;
        mx = fmaxf(mx, v);
    }
    red[tid] = mx; __syncthreads();
    for (int off = 128; off > 0; off >>= 1) {
        if (tid < off) red[tid] = fmaxf(red[tid], red[tid + off]);
        __syncthreads();
    }
    mx = red[0]; __syncthreads();
    float sum = 0.0f;
    for (int c = 0; c < cnt; ++c) { vals[c] = __expf(vals[c] - mx); sum += vals[c]; }
    red[tid] = sum; __syncthreads();
    for (int off = 128; off > 0; off >>= 1) {
        if (tid < off) red[tid] += red[tid + off];
        __syncthreads();
    }
    float inv = 1.0f / red[0];
    cnt = 0;
    for (int i = tid; i < rowlen; i += 256) row[i] = vals[cnt++] * inv;
}

__global__ __launch_bounds__(256)
void layernorm_kernel(const float* __restrict__ x, const float* __restrict__ g,
                      const float* __restrict__ b, const float* __restrict__ res,
                      float* __restrict__ y, int D)
{
    const float* xr = x + (long long)blockIdx.x * D;
    float* yr = y + (long long)blockIdx.x * D;
    const float* rr = res ? res + (long long)blockIdx.x * D : nullptr;
    __shared__ float red[256];
    int tid = threadIdx.x;
    float vals[5];
    int cnt = 0; float s = 0.0f;
    for (int i = tid; i < D; i += 256) { float v = xr[i]; vals[cnt++] = v; s += v; }
    red[tid] = s; __syncthreads();
    for (int off = 128; off > 0; off >>= 1) { if (tid < off) red[tid] += red[tid + off]; __syncthreads(); }
    float mean = red[0] / (float)D; __syncthreads();
    float s2 = 0.0f;
    for (int c = 0; c < cnt; ++c) { float dd = vals[c] - mean; s2 += dd * dd; }
    red[tid] = s2; __syncthreads();
    for (int off = 128; off > 0; off >>= 1) { if (tid < off) red[tid] += red[tid + off]; __syncthreads(); }
    float rstd = rsqrtf(red[0] / (float)D + 1e-5f);
    cnt = 0;
    for (int i = tid; i < D; i += 256) {
        float o = (vals[cnt++] - mean) * rstd * g[i] + b[i];
        if (rr) o += rr[i];
        yr[i] = o;
    }
}

// feats[m, k]: m over B*N points, k over 2d; layout [sin(x0 f), cos(x0 f), sin(x1 f), cos(x1 f)]
__global__ void feats_kernel(const float* __restrict__ x, float* __restrict__ feats, long long total)
{
    long long idx = (long long)blockIdx.x * blockDim.x + threadIdx.x;
    if (idx >= total) return;
    int m = (int)(idx / 2056);
    int k = (int)(idx - (long long)m * 2056);
    int coord = k / 1028;
    int r = k - coord * 1028;
    int hh = r / 514;
    int fi = r - hh * 514;
    float freq = __expf((2.0f * (float)fi) * (-9.210340371976184f / 1028.0f)); // ln(1e4)
    float v = x[(long long)m * 2 + coord] * freq;
    feats[idx] = hh ? cosf(v) : sinf(v);
}

__global__ void cls_kernel(const float* __restrict__ cls, float* __restrict__ h,
                           int d, int S, int B)
{
    int idx = blockIdx.x * blockDim.x + threadIdx.x;
    if (idx >= B * d) return;
    int b = idx / d, c = idx - b * d;
    h[(long long)b * S * d + c] = cls[c];
}

__global__ void head_kernel(const float* __restrict__ h, const float* __restrict__ w,
                            const float* __restrict__ bias, float* __restrict__ out,
                            int d, int S)
{
    int t = threadIdx.x;
    if (t >= 80) return;
    int b = t / 10, j = t - b * 10;
    const float* hr = h + (long long)b * S * d;
    float s = bias[j];
    for (int k = 0; k < d; ++k) s += hr[k] * w[k * 10 + j];
    out[b * 10 + j] = s;
}

extern "C" void kernel_launch(void* const* d_in, const int* in_sizes, int n_in,
                              void* d_out, int out_size, void* d_ws, size_t ws_size,
                              hipStream_t stream)
{
    (void)in_sizes; (void)n_in; (void)out_size; (void)ws_size;
    const int d = 1028, S = 2049, Spad = 2052, B = 8, Npts = 2048, d2 = 2056, d4 = 4112, L = 6;
    const int Mall = B * S; // 16392

    const float* x      = (const float*)d_in[0];
    const float* enc_w  = (const float*)d_in[1];
    const float* cls    = (const float*)d_in[2];
    const float* qw     = (const float*)d_in[3];
    const float* qbias  = (const float*)d_in[4];
    const float* kw     = (const float*)d_in[5];
    const float* kbias  = (const float*)d_in[6];
    const float* vw     = (const float*)d_in[7];
    const float* vbias  = (const float*)d_in[8];
    const float* ew     = (const float*)d_in[9];
    const float* eb     = (const float*)d_in[10];
    const float* ow     = (const float*)d_in[11];
    const float* ob     = (const float*)d_in[12];
    const float* ln_g   = (const float*)d_in[13];
    const float* ln_b   = (const float*)d_in[14];
    const float* out_w  = (const float*)d_in[15];
    const float* out_b  = (const float*)d_in[16];

    float* ws = (float*)d_ws;
    const size_t act = (size_t)Mall * d;           // 16,850,976 floats (mult of 4)
    float* h    = ws;            // persistent residual stream
    float* qb_  = h   + act;
    float* kb_  = qb_ + act;
    float* vb_  = kb_ + act;
    float* ab_  = vb_ + act;     // attention output
    float* tb_  = ab_ + act;     // post-LN1 (FFN input)
    float* fb_  = tb_ + act;     // FFN output
    float* big  = fb_ + act;     // shared: feats / scores(stride Spad) / ffn hidden

    dim3 blk(256);

    // 1) point-encoding features -> big (16384 x 2056)
    {
        long long tot = (long long)B * Npts * d2;
        feats_kernel<<<(unsigned)((tot + 255) / 256), blk, 0, stream>>>(x, big, tot);
    }
    // 2) CLS token into h row 0 of every batch
    cls_kernel<<<(B * d + 255) / 256, blk, 0, stream>>>(cls, h, d, S, B);
    // 3) encoder GEMM (per batch M=2048), writes h rows 1..2048
    {
        dim3 g((d + 127) / 128, (Npts + 127) / 128, B);
        gemm_kernel<0, false><<<g, blk, 0, stream>>>(
            big, d2, (long long)Npts * d2,
            enc_w, d, 0, nullptr,
            h + d, d, (long long)S * d,
            Npts, d, d2, 1.0f);
    }

    const float scale = 1.0f / sqrtf((float)d);

    for (int i = 0; i < L; ++i) {
        const float* wq = qw + (size_t)i * d * d;
        const float* wk = kw + (size_t)i * d * d;
        const float* wv = vw + (size_t)i * d * d;
        const float* we = ew + (size_t)i * d * d4;
        const float* wo = ow + (size_t)i * d4 * d;
        const float* bq = qbias + (size_t)i * d;
        const float* bk = kbias + (size_t)i * d;
        const float* bv = vbias + (size_t)i * d;
        const float* be = eb + (size_t)i * d4;
        const float* bo = ob + (size_t)i * d;
        const float* g_ = ln_g + (size_t)i * d;
        const float* b_ = ln_b + (size_t)i * d;

        // Q/K/V projections over flattened (B*S, d)
        dim3 gproj((d + 127) / 128, (Mall + 127) / 128, 1);
        gemm_kernel<0, false><<<gproj, blk, 0, stream>>>(h, d, 0, wq, d, 0, bq, qb_, d, 0, Mall, d, d, 1.0f);
        gemm_kernel<0, false><<<gproj, blk, 0, stream>>>(h, d, 0, wk, d, 0, bk, kb_, d, 0, Mall, d, d, 1.0f);
        gemm_kernel<0, false><<<gproj, blk, 0, stream>>>(h, d, 0, wv, d, 0, bv, vb_, d, 0, Mall, d, d, 1.0f);

        // scores = scale * Q @ K^T   (batched, BT path, padded row stride)
        dim3 gsc((S + 127) / 128, (S + 127) / 128, B);
        gemm_kernel<0, true><<<gsc, blk, 0, stream>>>(
            qb_, d, (long long)S * d,
            kb_, d, (long long)S * d, nullptr,
            big, Spad, (long long)S * Spad,
            S, S, d, scale);

        softmax_kernel<<<B * S, blk, 0, stream>>>(big, S, Spad);

        // attn = P @ V (batched)
        dim3 gav((d + 127) / 128, (S + 127) / 128, B);
        gemm_kernel<0, false><<<gav, blk, 0, stream>>>(
            big, Spad, (long long)S * Spad,
            vb_, d, (long long)S * d, nullptr,
            ab_, d, (long long)S * d,
            S, d, S, 1.0f);

        // LN1
        layernorm_kernel<<<Mall, blk, 0, stream>>>(ab_, g_, b_, nullptr, tb_, d);

        // FFN up + exact GELU
        dim3 gf1((d4 + 127) / 128, (Mall + 127) / 128, 1);
        gemm_kernel<1, false><<<gf1, blk, 0, stream>>>(tb_, d, 0, we, d4, 0, be, big, d4, 0, Mall, d4, d, 1.0f);
        // FFN down
        dim3 gf2((d + 127) / 128, (Mall + 127) / 128, 1);
        gemm_kernel<0, false><<<gf2, blk, 0, stream>>>(big, d4, 0, wo, d, 0, bo, fb_, d, 0, Mall, d, d4, 1.0f);

        // LN2 + residual add, h updated in place
        layernorm_kernel<<<Mall, blk, 0, stream>>>(fb_, g_, b_, h, h, d);
    }

    // classifier head on CLS tokens -> (8,10)
    head_kernel<<<1, 128, 0, stream>>>(h, out_w, out_b, (float*)d_out, d, S);
}
